// LatentSDE2_64785286693527
// MI455X (gfx1250) — matrix operand order repaired and checked by reading, not compiled, not run on hardware
//
#include <hip/hip_runtime.h>

typedef __attribute__((ext_vector_type(16))) __bf16 v16bf;
typedef __attribute__((ext_vector_type(8)))  __bf16 v8bf;
typedef __attribute__((ext_vector_type(8)))  float  v8f;

#define B_  512
#define T_  256
#define D_  32
#define L_  64
#define C_  128
#define H_  512
#define HE_ 256

static __device__ __forceinline__ float sigm(float x){ return 1.f/(1.f+__expf(-x)); }
static __device__ __forceinline__ float softplusf(float x){
  return x > 15.f ? x : __logf(1.f + __expf(x));
}

static constexpr float LOG2PI_F = 1.8378770664093453f;

// ---- workspace layout (bf16 elements) -------------------------------------
static constexpr size_t OFF_WXT = 0;                        // 768 x 64  (K padded 33->64)
static constexpr size_t OFF_WHT = OFF_WXT + 768*64;         // 768 x 256
static constexpr size_t OFF_ENCT= OFF_WHT + 768*256;        // 128 x 256
static constexpr size_t OFF_QZT = OFF_ENCT+ 128*256;        // 128 x 128
static constexpr size_t OFF_F1T = OFF_QZT + 128*128;        // 512 x 192
static constexpr size_t OFF_F2T = OFF_F1T + 512*192;        // 512 x 512
static constexpr size_t OFF_F3T = OFF_F2T + 512*512;        // 64  x 512
static constexpr size_t OFF_H1T = OFF_F3T + 64*512;         // 512 x 64
static constexpr size_t OFF_H2T = OFF_H1T + 512*64;         // 512 x 512
static constexpr size_t OFF_H3T = OFF_H2T + 512*512;        // 64  x 512
static constexpr size_t OFF_CTX = OFF_H3T + 64*512;         // T x B x C (bf16)

// ---- WMMA helpers ----------------------------------------------------------
static __device__ __forceinline__ v8f wmma_bf16(v16bf a, v16bf b, v8f c){
  return __builtin_amdgcn_wmma_f32_16x16x32_bf16(false, a, false, b, (short)0, c, false, false);
}

// A fragment (16x32 bf16, row-major source, row stride ld elements)
// lane: row = la (=lane&15), half = hi (=lane>>4)
// elems 0..7 : k = k0 + hi*8 + e    ; elems 8..15 : k = k0 + 16 + hi*8 + e
static __device__ __forceinline__ v16bf load_a(const __bf16* base, int ld, int k0, int la, int hi){
  const __bf16* p = base + (size_t)la*ld + k0 + hi*8;
  v8bf lo = *(const v8bf*)(p);
  v8bf h8 = *(const v8bf*)(p + 16);
  v16bf a;
#pragma unroll
  for (int e=0;e<8;e++){ a[e]=lo[e]; a[e+8]=h8[e]; }
  return a;
}

// B fragment (32x16 bf16) from transposed weights WT (N x K row-major):
// lane col n = nt*16+la ; elems 0..7 : k = k0 + hi*8 + e ; elems 8..15 : k = k0 + 16 + hi*8 + e
static __device__ __forceinline__ v16bf load_b(const __bf16* WT, int K, int nt, int k0, int la, int hi){
  const __bf16* p = WT + (size_t)(nt*16 + la)*K + k0 + hi*8;
  v8bf lo = *(const v8bf*)(p);
  v8bf h8 = *(const v8bf*)(p + 16);
  v16bf b;
#pragma unroll
  for (int e=0;e<8;e++){ b[e]=lo[e]; b[e+8]=h8[e]; }
  return b;
}

// generic dense layer: D = act(A @ W + b), A 16xK in LDS (bf16), out 16xN
template<int KTILES, int NT_PER_WAVE>
static __device__ __forceinline__ void mlp_layer(const __bf16* A, int lda,
    const __bf16* WT, int K, const float* bias,
    __bf16* dstb, float* dstf, int ldd, int wid, int la, int hi, bool soft)
{
#pragma unroll
  for (int i=0;i<NT_PER_WAVE;i++){
    int nt = wid*NT_PER_WAVE + i;
    v8f acc = {};
    for (int kt=0;kt<KTILES;kt++)
      acc = wmma_bf16(load_a(A, lda, kt*32, la, hi), load_b(WT, K, nt, kt*32, la, hi), acc);
    int cc = nt*16 + la;
    float bb = bias[cc];
#pragma unroll
    for (int v=0;v<8;v++){
      float r = acc[v] + bb;
      if (soft) r = softplusf(r);
      if (dstb) dstb[(v+8*hi)*ldd + cc] = (__bf16)r;
      else      dstf[(v+8*hi)*ldd + cc] = r;
    }
  }
}

// ---- weight transpose + fp32->bf16 conversion ------------------------------
__global__ void convert_wT(const float* __restrict__ src, __bf16* __restrict__ dst,
                           int fi, int fo, int Kpad){
  int idx = blockIdx.x*256 + threadIdx.x;
  int total = fo*Kpad;
  if (idx >= total) return;
  int n = idx / Kpad, k = idx - n*Kpad;
  dst[idx] = (k < fi) ? (__bf16)src[(size_t)k*fo + n] : (__bf16)0.0f;
}

// ---- fused persistent kernel: GRU scan -> q/z0 -> SDE scan -> outputs ------
struct SdeParams {
  const float *xs, *ts, *eps0, *dW;
  const float *gru_b, *enc_b, *qz0_b;
  const float *f_b1, *f_b2, *f_b3;
  const float *h_b1, *h_b2, *h_b3;
  const float *g_w1, *g_b1, *g_w2, *g_b2;
  const float *proj_W, *proj_b, *pz0_mean, *pz0_logstd;
  const __bf16 *WxT, *WhT, *encT, *qzT, *f1T, *f2T, *f3T, *h1T, *h2T, *h3T;
  __bf16 *ctx;   // [T][B][C] bf16
  float  *out;   // _xs (B,T,D) then logqp (B,1)
};

__global__ __launch_bounds__(256) void latent_sde_fused(SdeParams p){
  __shared__ __bf16 bufA[16*H_];     // 16 KB  activations
  __shared__ __bf16 bufB[16*H_];     // 16 KB
  __shared__ __bf16 hbuf[16*HE_];    // 8 KB   GRU state (bf16 for WMMA-A)
  __shared__ __bf16 zbf [16*L_];     // 2 KB   z state bf16
  __shared__ float  zf  [16*L_];     // 4 KB   z state fp32
  __shared__ float  fzb [16*L_];     // 4 KB
  __shared__ float  hzb [16*L_];     // 4 KB
  __shared__ float  projW[L_*D_];    // 8 KB
  __shared__ float  tsc[T_];         // 1 KB
  __shared__ float  dlogsum[16];
  __shared__ float  lq0[16];

  const int tid = threadIdx.x;
  const int la  = tid & 15;          // A-row / C-col within tile
  const int hi  = (tid >> 4) & 1;    // lane half
  const int wid = tid >> 5;          // wave id (8 waves)
  const int bbase = blockIdx.x * 16; // this WG's batch rows

  for (int i=tid;i<T_;i+=256)     tsc[i]   = p.ts[i];
  for (int i=tid;i<L_*D_;i+=256)  projW[i] = p.proj_W[i];
  for (int i=tid;i<16*HE_;i+=256) hbuf[i]  = (__bf16)0.0f;
  if (tid < 16){ dlogsum[tid]=0.f; lq0[tid]=0.f; }
  __syncthreads();

  // ================= Phase 1: GRU scan, emit ctx[t] to global ws ===========
#pragma unroll 1
  for (int t=0; t<T_; ++t){
    // x_in A-fragments: k<32 from xs, k==32 is ts[t], rest zero (Kpad=64)
    const float* xrow = p.xs + ((size_t)(bbase+la)*T_ + t)*D_ + hi*8;
    v16bf ax0;
#pragma unroll
    for (int e=0;e<8;e++){ ax0[e] = (__bf16)xrow[e]; ax0[e+8] = (__bf16)xrow[16+e]; }
    v16bf ax1;
#pragma unroll
    for (int e=0;e<16;e++) ax1[e] = (__bf16)0.0f;
    if (hi==0) ax1[0] = (__bf16)tsc[t];

    float hn[16];
#pragma unroll
    for (int jj=0;jj<2;jj++){
      int jc = wid + 8*jj;                       // gate column-tile (0..15)
      v8f ar={}, au={}, axn={}, ahn={};
      // x @ Wx  (r,u summed with hg later; n kept separate)
      ar  = wmma_bf16(ax0, load_b(p.WxT,64, jc,     0,la,hi), ar);
      ar  = wmma_bf16(ax1, load_b(p.WxT,64, jc,    32,la,hi), ar);
      au  = wmma_bf16(ax0, load_b(p.WxT,64, 16+jc,  0,la,hi), au);
      au  = wmma_bf16(ax1, load_b(p.WxT,64, 16+jc, 32,la,hi), au);
      axn = wmma_bf16(ax0, load_b(p.WxT,64, 32+jc,  0,la,hi), axn);
      axn = wmma_bf16(ax1, load_b(p.WxT,64, 32+jc, 32,la,hi), axn);
      // h @ Wh
      for (int kt=0;kt<8;kt++){
        v16bf ah = load_a(hbuf, HE_, kt*32, la, hi);
        ar  = wmma_bf16(ah, load_b(p.WhT,HE_, jc,    kt*32,la,hi), ar);
        au  = wmma_bf16(ah, load_b(p.WhT,HE_, 16+jc, kt*32,la,hi), au);
        ahn = wmma_bf16(ah, load_b(p.WhT,HE_, 32+jc, kt*32,la,hi), ahn);
      }
      int cc = jc*16 + la;
      float br = p.gru_b[cc], bu = p.gru_b[HE_+cc], bn = p.gru_b[2*HE_+cc];
#pragma unroll
      for (int v=0;v<8;v++){
        int row = v + 8*hi;
        float r  = sigm(ar[v]+br);
        float u  = sigm(au[v]+bu);
        float ng = tanhf(axn[v]+bn + r*ahn[v]);
        float ho = (float)hbuf[row*HE_ + cc];
        hn[jj*8+v] = (1.f-u)*ng + u*ho;
      }
    }
    __syncthreads();                              // all reads of h done
#pragma unroll
    for (int jj=0;jj<2;jj++){
      int cc = (wid+8*jj)*16 + la;
#pragma unroll
      for (int v=0;v<8;v++) hbuf[(v+8*hi)*HE_ + cc] = (__bf16)hn[jj*8+v];
    }
    __syncthreads();                              // h_new visible
    // ctx_t = h_new @ enc_W + enc_b  (wave w owns N-tile w; 128 cols)
    v8f ac = {};
    for (int kt=0;kt<8;kt++)
      ac = wmma_bf16(load_a(hbuf,HE_,kt*32,la,hi), load_b(p.encT,HE_,wid,kt*32,la,hi), ac);
    int cc = wid*16 + la;
    float be = p.enc_b[cc];
    __bf16* cd = p.ctx + ((size_t)t*B_ + bbase)*C_;
#pragma unroll
    for (int v=0;v<8;v++) cd[(v+8*hi)*C_ + cc] = (__bf16)(ac[v] + be);
  }
  __threadfence();
  __syncthreads();

  // ================= Phase 2: q, z0, logqp0, project t=0 ==================
  {
    const __bf16* c0 = p.ctx + (size_t)bbase * C_;          // ctx[0]
    float* qf = (float*)bufA;                               // 16x128 fp32 scratch
    mlp_layer<4,1>(c0, C_, p.qzT, C_, p.qz0_b, nullptr, qf, C_, wid, la, hi, false);
  }
  __syncthreads();
#pragma unroll
  for (int i=0;i<4;i++){
    int pp = tid + i*256; int mr = pp>>6; int l = pp&63;
    const float* qf = (const float*)bufA;
    float qm  = qf[mr*C_ + l];
    float qls = fminf(fmaxf(qf[mr*C_ + L_ + l], -20.f), 2.f);
    float z0  = qm + __expf(qls) * p.eps0[(size_t)(bbase+mr)*L_ + l];
    zf[pp] = z0; zbf[pp] = (__bf16)z0;
    float pm = p.pz0_mean[l], pls = p.pz0_logstd[l];
    float a1 = (z0-pm)*__expf(-pls);
    float a2 = (z0-qm)*__expf(-qls);
    float lp_p = -0.5f*a1*a1 - pls - 0.5f*LOG2PI_F;
    float lp_q = -0.5f*a2*a2 - qls - 0.5f*LOG2PI_F;
    atomicAdd(&lq0[mr], lp_p - lp_q);
  }
  __syncthreads();
  // _xs[:,0,:] = z0 @ proj_W + proj_b
#pragma unroll
  for (int i=0;i<2;i++){
    int q = tid*2+i; int mr = q>>5; int d = q&31;
    float s = p.proj_b[d];
    for (int l=0;l<L_;l++) s = fmaf(zf[mr*L_+l], projW[l*D_+d], s);
    p.out[((size_t)(bbase+mr)*T_)*D_ + d] = s;
  }

  // ================= Phase 3: SDE Euler scan (255 steps) ===================
  float dlacc[4] = {0.f,0.f,0.f,0.f};
#pragma unroll 1
  for (int t=0; t<T_-1; ++t){
    float dt = tsc[t+1]-tsc[t];
    float sq = sqrtf(dt);
    const __bf16* crow = p.ctx + ((size_t)(t+1)*B_ + bbase)*C_;

    // f layer1: [z | ctx] (K=192) -> H
#pragma unroll
    for (int i=0;i<4;i++){
      int nt = wid*4+i;
      v8f acc = {};
#pragma unroll
      for (int kt=0;kt<2;kt++)
        acc = wmma_bf16(load_a(zbf, L_, kt*32, la, hi), load_b(p.f1T,192,nt,kt*32,la,hi), acc);
#pragma unroll
      for (int kt=0;kt<4;kt++)
        acc = wmma_bf16(load_a(crow, C_, kt*32, la, hi), load_b(p.f1T,192,nt,(2+kt)*32,la,hi), acc);
      int cc = nt*16+la; float bb = p.f_b1[cc];
#pragma unroll
      for (int v=0;v<8;v++) bufA[(v+8*hi)*H_ + cc] = (__bf16)softplusf(acc[v]+bb);
    }
    __syncthreads();
    // f layer2
    mlp_layer<16,4>(bufA, H_, p.f2T, H_, p.f_b2, bufB, nullptr, H_, wid, la, hi, true);
    __syncthreads();
    // f layer3 (waves 0-3) -> fz ; h layer1 (all waves) -> bufA
    if (wid < 4)
      mlp_layer<16,1>(bufB, H_, p.f3T, H_, p.f_b3, nullptr, fzb, L_, wid, la, hi, false);
    mlp_layer<2,4>(zbf, L_, p.h1T, L_, p.h_b1, bufA, nullptr, H_, wid, la, hi, true);
    __syncthreads();
    // h layer2
    mlp_layer<16,4>(bufA, H_, p.h2T, H_, p.h_b2, bufB, nullptr, H_, wid, la, hi, true);
    __syncthreads();
    // h layer3 (waves 0-3) -> hz
    if (wid < 4)
      mlp_layer<16,1>(bufB, H_, p.h3T, H_, p.h_b3, nullptr, hzb, L_, wid, la, hi, false);
    __syncthreads();

    // diffusion g + state update (VALU transcendental-heavy part)
#pragma unroll
    for (int i=0;i<4;i++){
      int pp = tid + i*256; int mr = pp>>6; int l = pp&63;
      float zv = zf[pp];
      const float* w1 = p.g_w1 + (size_t)l*512;
      const float* b1 = p.g_b1 + (size_t)l*512;
      const float* w2 = p.g_w2 + (size_t)l*512;
      float acc = 0.f;
#pragma unroll 8
      for (int h=0; h<512; ++h)
        acc += softplusf(fmaf(zv, w1[h], b1[h])) * w2[h];
      float gz = sigm(acc + p.g_b2[l]);
      float u  = (fzb[pp] - hzb[pp]) / gz;
      dlacc[i] += 0.5f*u*u*dt;
      float dw = p.dW[((size_t)t*B_ + bbase + mr)*L_ + l];
      float zn = zv + fzb[pp]*dt + gz*sq*dw;
      zf[pp] = zn; zbf[pp] = (__bf16)zn;
    }
    __syncthreads();
    // _xs[:, t+1, :] = z_new @ proj_W + proj_b
#pragma unroll
    for (int i=0;i<2;i++){
      int q = tid*2+i; int mr = q>>5; int d = q&31;
      float s = p.proj_b[d];
      for (int l=0;l<L_;l++) s = fmaf(zf[mr*L_+l], projW[l*D_+d], s);
      p.out[((size_t)(bbase+mr)*T_ + (t+1))*D_ + d] = s;
    }
    __syncthreads();
  }

  // ================= Phase 4: logqp output =================================
#pragma unroll
  for (int i=0;i<4;i++) atomicAdd(&dlogsum[(tid + i*256)>>6], dlacc[i]);
  __syncthreads();
  if (tid < 16)
    p.out[(size_t)B_*T_*D_ + bbase + tid] = lq0[tid] - dlogsum[tid];
}

// ---- host launch -----------------------------------------------------------
extern "C" void kernel_launch(void* const* d_in, const int* in_sizes, int n_in,
                              void* d_out, int out_size, void* d_ws, size_t ws_size,
                              hipStream_t stream) {
  (void)in_sizes; (void)n_in; (void)out_size; (void)ws_size;
  const float* xs   = (const float*)d_in[0];
  const float* ts   = (const float*)d_in[1];
  const float* eps0 = (const float*)d_in[2];
  const float* dW   = (const float*)d_in[3];
  const float* gru_Wx = (const float*)d_in[4];
  const float* gru_Wh = (const float*)d_in[5];
  const float* gru_b  = (const float*)d_in[6];
  const float* enc_W  = (const float*)d_in[7];
  const float* enc_b  = (const float*)d_in[8];
  const float* qz0_W  = (const float*)d_in[9];
  const float* qz0_b  = (const float*)d_in[10];
  const float* f_W1 = (const float*)d_in[11];
  const float* f_b1 = (const float*)d_in[12];
  const float* f_W2 = (const float*)d_in[13];
  const float* f_b2 = (const float*)d_in[14];
  const float* f_W3 = (const float*)d_in[15];
  const float* f_b3 = (const float*)d_in[16];
  const float* h_W1 = (const float*)d_in[17];
  const float* h_b1 = (const float*)d_in[18];
  const float* h_W2 = (const float*)d_in[19];
  const float* h_b2 = (const float*)d_in[20];
  const float* h_W3 = (const float*)d_in[21];
  const float* h_b3 = (const float*)d_in[22];
  const float* g_w1 = (const float*)d_in[23];
  const float* g_b1 = (const float*)d_in[24];
  const float* g_w2 = (const float*)d_in[25];
  const float* g_b2 = (const float*)d_in[26];
  const float* proj_W = (const float*)d_in[27];
  const float* proj_b = (const float*)d_in[28];
  const float* pz0_mean   = (const float*)d_in[29];
  const float* pz0_logstd = (const float*)d_in[30];

  __bf16* ws = (__bf16*)d_ws;
  auto cvt = [&](const float* src, size_t off, int fi, int fo, int Kp){
    int total = fo*Kp;
    convert_wT<<<dim3((total+255)/256), dim3(256), 0, stream>>>(src, ws+off, fi, fo, Kp);
  };
  cvt(gru_Wx, OFF_WXT,  33, 768,  64);
  cvt(gru_Wh, OFF_WHT, 256, 768, 256);
  cvt(enc_W,  OFF_ENCT,256, 128, 256);
  cvt(qz0_W,  OFF_QZT, 128, 128, 128);
  cvt(f_W1,   OFF_F1T, 192, 512, 192);
  cvt(f_W2,   OFF_F2T, 512, 512, 512);
  cvt(f_W3,   OFF_F3T, 512,  64, 512);
  cvt(h_W1,   OFF_H1T,  64, 512,  64);
  cvt(h_W2,   OFF_H2T, 512, 512, 512);
  cvt(h_W3,   OFF_H3T, 512,  64, 512);

  SdeParams prm;
  prm.xs = xs; prm.ts = ts; prm.eps0 = eps0; prm.dW = dW;
  prm.gru_b = gru_b; prm.enc_b = enc_b; prm.qz0_b = qz0_b;
  prm.f_b1 = f_b1; prm.f_b2 = f_b2; prm.f_b3 = f_b3;
  prm.h_b1 = h_b1; prm.h_b2 = h_b2; prm.h_b3 = h_b3;
  prm.g_w1 = g_w1; prm.g_b1 = g_b1; prm.g_w2 = g_w2; prm.g_b2 = g_b2;
  prm.proj_W = proj_W; prm.proj_b = proj_b;
  prm.pz0_mean = pz0_mean; prm.pz0_logstd = pz0_logstd;
  prm.WxT = ws+OFF_WXT; prm.WhT = ws+OFF_WHT; prm.encT = ws+OFF_ENCT; prm.qzT = ws+OFF_QZT;
  prm.f1T = ws+OFF_F1T; prm.f2T = ws+OFF_F2T; prm.f3T = ws+OFF_F3T;
  prm.h1T = ws+OFF_H1T; prm.h2T = ws+OFF_H2T; prm.h3T = ws+OFF_H3T;
  prm.ctx = ws+OFF_CTX;
  prm.out = (float*)d_out;

  latent_sde_fused<<<dim3(B_/16), dim3(256), 0, stream>>>(prm);
}
